// GeneralRNN_8555574853748
// MI455X (gfx1250) — matrix operand order repaired and checked by reading, not compiled, not run on hardware
//
#include <hip/hip_runtime.h>
#include <hip/hip_bf16.h>

// ---------------- problem constants ----------------
#define BDIM 128
#define TLEN 2048
#define DIN  64
#define HDIM 128
#define WH   256
#define WY   256
#define DOUT 64
#define KHX  192   // HDIM + DIN

// ---------------- WMMA types ----------------
typedef __attribute__((ext_vector_type(16))) __bf16 v16bf;
typedef __attribute__((ext_vector_type(8)))  float  v8f;

union FragU {
    v16bf bf;
    uint4 u[2];
};

// ---------------- LDS layout (bytes) ----------------
static constexpr int OFF_HW1 = 0;                       // [256][192] bf16 = 98304
static constexpr int OFF_HW2 = OFF_HW1 + WH * KHX * 2;  // [128][256] bf16 = 65536
static constexpr int OFF_YW1 = OFF_HW2 + HDIM * WH * 2; // [256][128] bf16 = 65536
static constexpr int OFF_YW2 = OFF_YW1 + WY * HDIM * 2; // [ 64][256] bf16 = 32768
static constexpr int OFF_HX  = OFF_YW2 + DOUT * WY * 2; // [16][192]  bf16 =  6144
static constexpr int OFF_A1  = OFF_HX  + 16 * KHX * 2;  // [16][256]  bf16 =  8192
static constexpr int OFF_A2  = OFF_A1  + 16 * WH * 2;   // [16][256]  bf16 =  8192
static constexpr int OFF_HB1 = OFF_A2  + 16 * WY * 2;   // 256 f32
static constexpr int OFF_HB2 = OFF_HB1 + WH * 4;        // 128 f32
static constexpr int OFF_YB1 = OFF_HB2 + HDIM * 4;      // 256 f32
static constexpr int OFF_YB2 = OFF_YB1 + WY * 4;        //  64 f32
static constexpr int SMEM_BYTES = OFF_YB2 + DOUT * 4;   // = 287488 (< 320K WGP LDS)

// ---------------- helpers ----------------
__device__ __forceinline__ v8f splat8(float v) {
    v8f r;
#pragma unroll
    for (int i = 0; i < 8; ++i) r[i] = v;
    return r;
}

__device__ __forceinline__ v8f bfwmma(v16bf a, v16bf b, v8f c) {
    // D = A * B + C, f32 accumulate
    return __builtin_amdgcn_wmma_f32_16x16x32_bf16(
        /*neg_a=*/false, a, /*neg_b=*/false, b,
        /*c_mod=*/(short)0, c, /*reuse_a=*/false, /*reuse_b=*/false);
}

// A fragment, 16x32 bf16 (ISA 7.12.2 "16-bit A-Matrix 16x32"):
// lane l: row m = l&15, half = l>>4; VGPR r holds packed K pair at
//   K = 16*(r>>2) + 8*half + 2*(r&3)  => two ds_load_b128 at +0 and +16 elems.
__device__ __forceinline__ v16bf ldsFragA(const __bf16* base, int row, int ld,
                                          int k0, int lane) {
    const int half8 = (lane >> 4) << 3;              // 0 or 8
    const __bf16* p = base + row * ld + k0 + half8;  // 16B aligned
    FragU f;
    f.u[0] = *(const uint4*)(const void*)(p);        // K group  0..15 side
    f.u[1] = *(const uint4*)(const void*)(p + 16);   // K group 16..31 side
    return f.bf;
}

// B fragment, 32x16 bf16 (ISA 7.12.4 B-matrix pattern, scaled to dense K=32):
// same K order as A: VGPR r holds packed pair
// K = 16*(r>>2) + 8*half + 2*(r&3). With W stored [N][K] row-major and
// B[k][n] = W[n][k], that is two 16B loads of W's row n at +0 and +16 elems.
__device__ __forceinline__ v16bf ldsFragB(const __bf16* base, int n, int ld,
                                          int k0, int lane) {
    const int khalf = (lane >> 4) << 3;              // 0 or 8
    const __bf16* p = base + n * ld + k0 + khalf;    // 16B aligned
    FragU f;
    f.u[0] = *(const uint4*)(const void*)(p);        // K = khalf + 0..7
    f.u[1] = *(const uint4*)(const void*)(p + 16);   // K = 16 + khalf + 0..7
    return f.bf;
}

__device__ __forceinline__ float silu(float a) {
    return a / (1.0f + __expf(-a));
}

__device__ __forceinline__ void cvt_store_bf16(const float* __restrict__ src,
                                               __bf16* dst, int n, int tid) {
    for (int i = tid; i < n; i += 256) dst[i] = (__bf16)src[i];
}

// ---------------- persistent RNN kernel ----------------
__global__ __launch_bounds__(256, 1)
void rnn_scan_kernel(const float* __restrict__ x,
                     const float* __restrict__ hW1, const float* __restrict__ hb1,
                     const float* __restrict__ hW2, const float* __restrict__ hb2,
                     const float* __restrict__ yW1, const float* __restrict__ yb1,
                     const float* __restrict__ yW2, const float* __restrict__ yb2,
                     float* __restrict__ out,   // [B, T, DOUT]
                     float* __restrict__ hid) { // [B, HDIM]
    __shared__ __attribute__((aligned(16))) unsigned char smem[SMEM_BYTES];

    __bf16* sHW1 = (__bf16*)(smem + OFF_HW1);
    __bf16* sHW2 = (__bf16*)(smem + OFF_HW2);
    __bf16* sYW1 = (__bf16*)(smem + OFF_YW1);
    __bf16* sYW2 = (__bf16*)(smem + OFF_YW2);
    __bf16* sHX  = (__bf16*)(smem + OFF_HX);
    __bf16* sA1  = (__bf16*)(smem + OFF_A1);
    __bf16* sA2  = (__bf16*)(smem + OFF_A2);
    float*  sHB1 = (float*)(smem + OFF_HB1);
    float*  sHB2 = (float*)(smem + OFF_HB2);
    float*  sYB1 = (float*)(smem + OFF_YB1);
    float*  sYB2 = (float*)(smem + OFF_YB2);

    const int tid   = threadIdx.x;
    const int lane  = tid & 31;
    const int wave  = tid >> 5;          // 0..7
    const int nrow  = lane & 15;         // M/N row within tile
    const int half  = lane >> 4;         // 0/1
    const int bbase = blockIdx.x * 16;   // 8 blocks x 16 batch rows

    // ---- one-time: stage all weights to LDS as bf16, biases as f32 ----
    cvt_store_bf16(hW1, sHW1, WH * KHX, tid);
    cvt_store_bf16(hW2, sHW2, HDIM * WH, tid);
    cvt_store_bf16(yW1, sYW1, WY * HDIM, tid);
    cvt_store_bf16(yW2, sYW2, DOUT * WY, tid);
    for (int i = tid; i < WH;   i += 256) sHB1[i] = hb1[i];
    for (int i = tid; i < HDIM; i += 256) sHB2[i] = hb2[i];
    for (int i = tid; i < WY;   i += 256) sYB1[i] = yb1[i];
    for (int i = tid; i < DOUT; i += 256) sYB2[i] = yb2[i];
    // h0 = 0 in hx[:, 0:128]
    for (int i = tid; i < 16 * HDIM; i += 256) {
        int m = i >> 7, c = i & 127;
        sHX[m * KHX + c] = (__bf16)0.0f;
    }
    __syncthreads();

    for (int t = 0; t < TLEN; ++t) {
        // ---- stage xt -> hx[:, 128:192] (one float4 per thread) ----
        {
            const int m = tid >> 4;            // 0..15
            const int d = (tid & 15) * 4;      // 0..60
            const size_t gbase = ((size_t)(bbase + m) * TLEN + t) * DIN + d;
            const float4 xv = *(const float4*)(const void*)(x + gbase);
            __bf16* dst = sHX + m * KHX + HDIM + d;
            dst[0] = (__bf16)xv.x; dst[1] = (__bf16)xv.y;
            dst[2] = (__bf16)xv.z; dst[3] = (__bf16)xv.w;
            if (t + 1 < TLEN)  // global_prefetch_b8 for next step's xt
                __builtin_prefetch(x + gbase + DIN, 0, 1);
        }
        __syncthreads();

        // ---- layer 1: act1 = silu(hx @ hW1^T + hb1)   [16,192]x[192,256] ----
        {
            const int n0 = wave * 32 + nrow;
            const int n1 = n0 + 16;
            v8f acc0 = splat8(sHB1[n0]);
            v8f acc1 = splat8(sHB1[n1]);
#pragma unroll
            for (int kk = 0; kk < 6; ++kk) {
                v16bf a  = ldsFragA(sHX,  nrow, KHX, kk * 32, lane);
                v16bf w0 = ldsFragB(sHW1, n0,   KHX, kk * 32, lane);
                v16bf w1 = ldsFragB(sHW1, n1,   KHX, kk * 32, lane);
                acc0 = bfwmma(a, w0, acc0);
                acc1 = bfwmma(a, w1, acc1);
            }
#pragma unroll
            for (int j = 0; j < 8; ++j) {
                const int m = j + 8 * half;
                sA1[m * WH + n0] = (__bf16)silu(acc0[j]);
                sA1[m * WH + n1] = (__bf16)silu(acc1[j]);
            }
        }
        __syncthreads();

        // ---- layer 2: h_new = act1 @ hW2^T + hb2   [16,256]x[256,128] ----
        {
            const int n = wave * 16 + nrow;   // 0..127
            v8f acc = splat8(sHB2[n]);
#pragma unroll
            for (int kk = 0; kk < 8; ++kk) {
                v16bf a = ldsFragA(sA1,  nrow, WH, kk * 32, lane);
                v16bf w = ldsFragB(sHW2, n,    WH, kk * 32, lane);
                acc = bfwmma(a, w, acc);
            }
#pragma unroll
            for (int j = 0; j < 8; ++j) {
                const int m = j + 8 * half;
                sHX[m * KHX + n] = (__bf16)acc[j];   // recurrent state, bf16
            }
            if (t == TLEN - 1) {  // final hidden, f32
#pragma unroll
                for (int j = 0; j < 8; ++j) {
                    const int m = j + 8 * half;
                    hid[(size_t)(bbase + m) * HDIM + n] = acc[j];
                }
            }
        }
        __syncthreads();

        // ---- layer 3: act2 = silu(h_new @ yW1^T + yb1)  [16,128]x[128,256] ----
        {
            const int n0 = wave * 32 + nrow;
            const int n1 = n0 + 16;
            v8f acc0 = splat8(sYB1[n0]);
            v8f acc1 = splat8(sYB1[n1]);
#pragma unroll
            for (int kk = 0; kk < 4; ++kk) {
                v16bf a  = ldsFragA(sHX,  nrow, KHX,  kk * 32, lane); // K in [0,128)
                v16bf w0 = ldsFragB(sYW1, n0,   HDIM, kk * 32, lane);
                v16bf w1 = ldsFragB(sYW1, n1,   HDIM, kk * 32, lane);
                acc0 = bfwmma(a, w0, acc0);
                acc1 = bfwmma(a, w1, acc1);
            }
#pragma unroll
            for (int j = 0; j < 8; ++j) {
                const int m = j + 8 * half;
                sA2[m * WY + n0] = (__bf16)silu(acc0[j]);
                sA2[m * WY + n1] = (__bf16)silu(acc1[j]);
            }
        }
        __syncthreads();

        // ---- layer 4: y = act2 @ yW2^T + yb2   [16,256]x[256,64] ----
        // No trailing barrier needed: L4's sA2 reads are ordered before the
        // stage-1 barrier of step t+1, and the next sA2 writer (L3 of t+1)
        // sits behind two more barriers.
        if (wave < 4) {
            const int n = wave * 16 + nrow;   // 0..63
            v8f acc = splat8(sYB2[n]);
#pragma unroll
            for (int kk = 0; kk < 8; ++kk) {
                v16bf a = ldsFragA(sA2,  nrow, WY, kk * 32, lane);
                v16bf w = ldsFragB(sYW2, n,    WY, kk * 32, lane);
                acc = bfwmma(a, w, acc);
            }
#pragma unroll
            for (int j = 0; j < 8; ++j) {
                const int m = j + 8 * half;
                out[((size_t)(bbase + m) * TLEN + t) * DOUT + n] = acc[j];
            }
        }
    }
}

// ---------------- launcher ----------------
extern "C" void kernel_launch(void* const* d_in, const int* in_sizes, int n_in,
                              void* d_out, int out_size, void* d_ws, size_t ws_size,
                              hipStream_t stream) {
    (void)in_sizes; (void)n_in; (void)out_size; (void)d_ws; (void)ws_size;
    const float* x   = (const float*)d_in[0];
    const float* hW1 = (const float*)d_in[1];
    const float* hb1 = (const float*)d_in[2];
    const float* hW2 = (const float*)d_in[3];
    const float* hb2 = (const float*)d_in[4];
    const float* yW1 = (const float*)d_in[5];
    const float* yb1 = (const float*)d_in[6];
    const float* yW2 = (const float*)d_in[7];
    const float* yb2 = (const float*)d_in[8];

    float* out = (float*)d_out;
    float* hid = out + (size_t)BDIM * TLEN * DOUT;  // tuple: (out, hidden) flat

    hipLaunchKernelGGL(rnn_scan_kernel, dim3(BDIM / 16), dim3(256), 0, stream,
                       x, hW1, hb1, hW2, hb2, yW1, yb1, yW2, yb2, out, hid);
}
